// LlmcDeepseekAttention_5841155522793
// MI455X (gfx1250) — matrix operand order repaired and checked by reading, not compiled, not run on hardware
//
#include <hip/hip_runtime.h>
#include <hip/hip_bf16.h>

// ---------------- problem constants ----------------
#define B_     2
#define S_     2048
#define HID_   5120
#define H_     16
#define QLORA  1536
#define KVLORA 512
#define DROPE  64
#define DNOPE  128
#define DV     128
#define DQ     192            // DNOPE + DROPE
#define TOK    (B_ * S_)      // 4096 tokens

typedef __attribute__((ext_vector_type(16))) __bf16 v16bf;
typedef __attribute__((ext_vector_type(8)))  __bf16 v8bf;
typedef __attribute__((ext_vector_type(8)))  float  v8f;

union frag16 { v16bf v; v8bf h[2]; };

// f32 -> bf16 via native conversion; backend picks v_cvt_pk_bf16_f32-class
// ops on gfx1250 (falls back to inline expansion elsewhere, still compiles).
__device__ inline __bf16 f2bf(float f) { return (__bf16)f; }

// =====================================================================
// GEMM: C[M,N] = A[M,K] @ W[N,K]^T   (f32 in/out, bf16 WMMA compute)
// Block: 128 threads = 4 waves. Block tile 128x64, K step 32.
// Each wave computes a 32x64 subtile = 2x4 WMMA 16x16 tiles
// (8 WMMA per 12 ds_load_b128 per K-step -> matrix-pipe friendly).
// =====================================================================
__global__ __launch_bounds__(128)
void gemm_wmma_bf16(const float* __restrict__ A, const float* __restrict__ W,
                    float* __restrict__ C, int M, int N, int K) {
  __shared__ __bf16 As[128 * 40];  // 128 rows x 32 K, stride 40 (16B-aligned rows)
  __shared__ __bf16 Bs[64 * 40];   //  64 rows x 32 K

  const int tid  = threadIdx.x;
  const int wave = tid >> 5;
  const int lane = tid & 31;
  const int lo   = lane & 15;
  const int hi   = lane >> 4;
  const int n0   = blockIdx.x * 64;
  const int m0   = blockIdx.y * 128;
  const int mw   = wave * 32;       // wave's M offset inside the block tile

  v8f acc[2][4] = {};

  const int ar  = tid;              // A staging: thread -> row, full 32 K cols
  const int br  = tid >> 1;         // B staging: 2 threads per row
  const int bc0 = (tid & 1) * 16;

  for (int k0 = 0; k0 < K; k0 += 32) {
    __syncthreads();
    // cooperative stage of A/W tiles into LDS as bf16 (native cvt)
    {
      const float4* ap = (const float4*)(A + (size_t)(m0 + ar) * K + k0);
#pragma unroll
      for (int q4 = 0; q4 < 8; ++q4) {
        float4 av = ap[q4];
        int base = ar * 40 + q4 * 4;
        As[base + 0] = f2bf(av.x); As[base + 1] = f2bf(av.y);
        As[base + 2] = f2bf(av.z); As[base + 3] = f2bf(av.w);
      }
      const float4* bp = (const float4*)(W + (size_t)(n0 + br) * K + k0 + bc0);
#pragma unroll
      for (int q4 = 0; q4 < 4; ++q4) {
        float4 bv = bp[q4];
        int base = br * 40 + bc0 + q4 * 4;
        Bs[base + 0] = f2bf(bv.x); Bs[base + 1] = f2bf(bv.y);
        Bs[base + 2] = f2bf(bv.z); Bs[base + 3] = f2bf(bv.w);
      }
    }
    if (k0 + 32 < K) {
      __builtin_prefetch(A + (size_t)(m0 + ar) * K + k0 + 32, 0, 1);
      __builtin_prefetch(W + (size_t)(n0 + br) * K + k0 + 32 + bc0, 0, 1);
    }
    __syncthreads();

    // fragment loads per ISA wave32 VGPR layouts
    frag16 afr[2], bfr[4];
#pragma unroll
    for (int mi = 0; mi < 2; ++mi) {
      int row = mw + mi * 16 + lo;
      int kb  = hi * 8;              // A: elems 0-7 -> K kb..kb+7, 8-15 -> kb+16..kb+23
      afr[mi].h[0] = *(const v8bf*)&As[row * 40 + kb];
      afr[mi].h[1] = *(const v8bf*)&As[row * 40 + kb + 16];
    }
#pragma unroll
    for (int ni = 0; ni < 4; ++ni) {
      int row = ni * 16 + lo;
      int kb  = hi * 8;              // B: elems 0-7 -> K kb..kb+7, 8-15 -> kb+16..kb+23
      bfr[ni].h[0] = *(const v8bf*)&Bs[row * 40 + kb];
      bfr[ni].h[1] = *(const v8bf*)&Bs[row * 40 + kb + 16];
    }
#pragma unroll
    for (int mi = 0; mi < 2; ++mi)
#pragma unroll
      for (int ni = 0; ni < 4; ++ni)
        acc[mi][ni] = __builtin_amdgcn_wmma_f32_16x16x32_bf16(
            false, afr[mi].v, false, bfr[ni].v, (short)0, acc[mi][ni], false, false);
  }

  // epilogue: C layout M = rr + 8*hi, N = lo
#pragma unroll
  for (int mi = 0; mi < 2; ++mi)
#pragma unroll
    for (int ni = 0; ni < 4; ++ni) {
      int col = n0 + ni * 16 + lo;
#pragma unroll
      for (int rr = 0; rr < 8; ++rr) {
        int row = m0 + mw + mi * 16 + rr + 8 * hi;
        C[(size_t)row * N + col] = acc[mi][ni][rr];
      }
    }
}

// =====================================================================
// RMS norm (row-local). One block of 256 threads per row.
// =====================================================================
__global__ __launch_bounds__(256)
void rmsnorm_kernel(const float* __restrict__ x, const float* __restrict__ w,
                    float* __restrict__ y, int in_stride, int out_stride, int D) {
  __shared__ float wsum[8];
  const int row = blockIdx.x;
  const int tid = threadIdx.x;
  const float* xr = x + (size_t)row * in_stride;
  float s = 0.f;
  for (int i = tid; i < D; i += 256) { float v = xr[i]; s += v * v; }
  for (int off = 16; off > 0; off >>= 1) s += __shfl_xor(s, off, 32);
  if ((tid & 31) == 0) wsum[tid >> 5] = s;
  __syncthreads();
  if (tid < 8) {
    float t = wsum[tid];
    for (int off = 4; off > 0; off >>= 1) t += __shfl_xor(t, off, 32);
    if (tid == 0) wsum[0] = t;
  }
  __syncthreads();
  float inv = rsqrtf(wsum[0] / (float)D + 1e-6f);
  float* yr = y + (size_t)row * out_stride;
  for (int i = tid; i < D; i += 256) yr[i] = xr[i] * inv * w[i];
}

// =====================================================================
// RoPE with deinterleave, applied in-place to q_pe (all heads) and
// producing k_pe from ckv tail. Two-phase (read all / barrier / write all)
// because the deinterleave is an in-place permutation.
// out[j]    = x[2j]*cos_j   - x[2j+1]*sin_j
// out[32+j] = x[2j+1]*cos_j + x[2j]*sin_j
// =====================================================================
__global__ __launch_bounds__(256)
void rope_kernel(float* __restrict__ qb, const float* __restrict__ ckv,
                 float* __restrict__ kpe, const float* __restrict__ cosb,
                 const float* __restrict__ sinb) {
  const int t = blockIdx.x;
  const int pos = t & (S_ - 1);
  const int nitems = H_ * 32 + 32;   // 16 heads * 32 pairs + 32 k_pe pairs
  float r0[3], r1[3];
#pragma unroll
  for (int p = 0; p < 3; ++p) {
    int i = threadIdx.x + p * 256;
    if (i < nitems) {
      int j  = i & 31;
      float c  = cosb[pos * DROPE + j];
      float sn = sinb[pos * DROPE + j];
      float x0, x1;
      if (i < H_ * 32) {
        int h = i >> 5;
        const float* qp = qb + (size_t)t * (H_ * DQ) + h * DQ + DNOPE;
        x0 = qp[2 * j]; x1 = qp[2 * j + 1];
      } else {
        const float* kp = ckv + (size_t)t * (KVLORA + DROPE) + KVLORA;
        x0 = kp[2 * j]; x1 = kp[2 * j + 1];
      }
      r0[p] = x0 * c - x1 * sn;
      r1[p] = x1 * c + x0 * sn;
    }
  }
  __syncthreads();
#pragma unroll
  for (int p = 0; p < 3; ++p) {
    int i = threadIdx.x + p * 256;
    if (i < nitems) {
      int j = i & 31;
      if (i < H_ * 32) {
        int h = i >> 5;
        float* qp = qb + (size_t)t * (H_ * DQ) + h * DQ + DNOPE;
        qp[j] = r0[p]; qp[32 + j] = r1[p];
      } else {
        kpe[(size_t)t * DROPE + j]      = r0[p];
        kpe[(size_t)t * DROPE + 32 + j] = r1[p];
      }
    }
  }
}

// =====================================================================
// Flash attention, causal. Grid: (S/64 q-tiles, B*H). Block: 4 waves.
// Each wave owns 16 query rows. K and V^T staged in LDS as bf16 per
// 64-key tile; online softmax in f32; P goes through LDS to reach the
// WMMA A-fragment layout for the PV product.
// =====================================================================
#define QT 64
__global__ __launch_bounds__(128)
void attn_kernel(const float* __restrict__ q, const float* __restrict__ kvbuf,
                 const float* __restrict__ kpe, float* __restrict__ out) {
  __shared__ __bf16 Ks[64 * 200];   // 64 keys x 192 d (stride 200)
  __shared__ __bf16 Vt[128 * 72];   // 128 dv  x 64 keys (transposed, stride 72)
  __shared__ __bf16 Ps[64 * 72];    // 64 q    x 64 keys (stride 72)

  const int tid  = threadIdx.x;
  const int wave = tid >> 5;
  const int lane = tid & 31;
  const int lo   = lane & 15;
  const int hi   = lane >> 4;
  const int qt   = blockIdx.x;
  const int bh   = blockIdx.y;
  const int b    = bh >> 4;
  const int h    = bh & 15;
  const int tb   = b * S_ + qt * QT;   // token base of this query tile

  const float scale = 0.07216878364870323f;   // 1/sqrt(192)

  // Q fragments (scaled), 6 K-steps of 32 over d=192
  frag16 qf[6];
  {
    int tok = tb + wave * 16 + lo;
    const float* qrow = q + (size_t)tok * (H_ * DQ) + h * DQ;
#pragma unroll
    for (int s6 = 0; s6 < 6; ++s6) {
      int kb = s6 * 32 + hi * 8;
#pragma unroll
      for (int g = 0; g < 2; ++g) {
        const float4* p = (const float4*)(qrow + kb + g * 16);
        float4 a = p[0], c = p[1];
        qf[s6].h[g][0] = f2bf(a.x * scale); qf[s6].h[g][1] = f2bf(a.y * scale);
        qf[s6].h[g][2] = f2bf(a.z * scale); qf[s6].h[g][3] = f2bf(a.w * scale);
        qf[s6].h[g][4] = f2bf(c.x * scale); qf[s6].h[g][5] = f2bf(c.y * scale);
        qf[s6].h[g][6] = f2bf(c.z * scale); qf[s6].h[g][7] = f2bf(c.w * scale);
      }
    }
  }

  v8f oacc[8] = {};
  float m_i[8], l_i[8];
#pragma unroll
  for (int rr = 0; rr < 8; ++rr) { m_i[rr] = -1e30f; l_i[rr] = 0.f; }

  for (int kt = 0; kt <= qt; ++kt) {
    const int kb0 = b * S_ + kt * QT;
    __syncthreads();
    // stage K tile (k_nope ++ roped k_pe) 64x192
    for (int i = tid; i < 64 * 48; i += 128) {
      int rr = i / 48, c4 = i % 48, col = c4 * 4;
      float4 v;
      if (col < DNOPE)
        v = *(const float4*)(kvbuf + (size_t)(kb0 + rr) * (H_ * 256) + h * 256 + col);
      else
        v = *(const float4*)(kpe + (size_t)(kb0 + rr) * DROPE + (col - DNOPE));
      int base = rr * 200 + col;
      Ks[base + 0] = f2bf(v.x); Ks[base + 1] = f2bf(v.y);
      Ks[base + 2] = f2bf(v.z); Ks[base + 3] = f2bf(v.w);
    }
    // stage V^T tile 128x64
    for (int i = tid; i < 64 * 32; i += 128) {
      int rr = i / 32, c4 = i % 32, col = c4 * 4;
      float4 v = *(const float4*)(kvbuf + (size_t)(kb0 + rr) * (H_ * 256) + h * 256 + 128 + col);
      Vt[(col + 0) * 72 + rr] = f2bf(v.x);
      Vt[(col + 1) * 72 + rr] = f2bf(v.y);
      Vt[(col + 2) * 72 + rr] = f2bf(v.z);
      Vt[(col + 3) * 72 + rr] = f2bf(v.w);
    }
    __syncthreads();

    // S = (Q*scale) K^T, 16 q-rows x 64 keys per wave
    v8f sacc[4] = {};
#pragma unroll
    for (int ni = 0; ni < 4; ++ni) {
#pragma unroll
      for (int s6 = 0; s6 < 6; ++s6) {
        frag16 kf;
        int krow = ni * 16 + lo;
        int kb   = s6 * 32 + hi * 8;
        kf.h[0] = *(const v8bf*)&Ks[krow * 200 + kb];
        kf.h[1] = *(const v8bf*)&Ks[krow * 200 + kb + 16];
        sacc[ni] = __builtin_amdgcn_wmma_f32_16x16x32_bf16(
            false, qf[s6].v, false, kf.v, (short)0, sacc[ni], false, false);
      }
    }
    // causal mask (diagonal tile only)
    if (kt == qt) {
#pragma unroll
      for (int ni = 0; ni < 4; ++ni) {
        int kg = kt * QT + ni * 16 + lo;
#pragma unroll
        for (int rr = 0; rr < 8; ++rr) {
          int qg = qt * QT + wave * 16 + rr + 8 * hi;
          if (kg > qg) sacc[ni][rr] += -1e9f;
        }
      }
    }
    // online softmax (rows live in 16-lane half-wave groups)
    float tmax[8];
#pragma unroll
    for (int rr = 0; rr < 8; ++rr) tmax[rr] = -1e30f;
#pragma unroll
    for (int ni = 0; ni < 4; ++ni)
#pragma unroll
      for (int rr = 0; rr < 8; ++rr) tmax[rr] = fmaxf(tmax[rr], sacc[ni][rr]);
#pragma unroll
    for (int rr = 0; rr < 8; ++rr)
      for (int off = 1; off < 16; off <<= 1)
        tmax[rr] = fmaxf(tmax[rr], __shfl_xor(tmax[rr], off, 32));
    float alpha[8];
#pragma unroll
    for (int rr = 0; rr < 8; ++rr) {
      float mn = fmaxf(m_i[rr], tmax[rr]);
      alpha[rr] = __expf(m_i[rr] - mn);
      m_i[rr] = mn;
    }
    float rsum[8] = {};
#pragma unroll
    for (int ni = 0; ni < 4; ++ni)
#pragma unroll
      for (int rr = 0; rr < 8; ++rr) {
        float p = __expf(sacc[ni][rr] - m_i[rr]);
        sacc[ni][rr] = p;
        rsum[rr] += p;
      }
#pragma unroll
    for (int rr = 0; rr < 8; ++rr)
      for (int off = 1; off < 16; off <<= 1)
        rsum[rr] += __shfl_xor(rsum[rr], off, 32);
#pragma unroll
    for (int rr = 0; rr < 8; ++rr) l_i[rr] = l_i[rr] * alpha[rr] + rsum[rr];
#pragma unroll
    for (int nt = 0; nt < 8; ++nt)
#pragma unroll
      for (int rr = 0; rr < 8; ++rr) oacc[nt][rr] *= alpha[rr];

    // P -> LDS (C layout -> memory), then reload in A-fragment layout
#pragma unroll
    for (int ni = 0; ni < 4; ++ni)
#pragma unroll
      for (int rr = 0; rr < 8; ++rr)
        Ps[(wave * 16 + rr + 8 * hi) * 72 + ni * 16 + lo] = f2bf(sacc[ni][rr]);
    __syncthreads();

    // O += P @ V  (inner dim = 64 keys, 2 steps of 32)
#pragma unroll
    for (int ks = 0; ks < 2; ++ks) {
      frag16 pf;
      int prow = wave * 16 + lo;
      int kb   = ks * 32 + hi * 8;
      pf.h[0] = *(const v8bf*)&Ps[prow * 72 + kb];
      pf.h[1] = *(const v8bf*)&Ps[prow * 72 + kb + 16];
#pragma unroll
      for (int nt = 0; nt < 8; ++nt) {
        frag16 vf;
        int dv  = nt * 16 + lo;
        int kb2 = ks * 32 + hi * 8;
        vf.h[0] = *(const v8bf*)&Vt[dv * 72 + kb2];
        vf.h[1] = *(const v8bf*)&Vt[dv * 72 + kb2 + 16];
        oacc[nt] = __builtin_amdgcn_wmma_f32_16x16x32_bf16(
            false, pf.v, false, vf.v, (short)0, oacc[nt], false, false);
      }
    }
  }

  // finalize: out[token][h*128 + dv] = O / l
#pragma unroll
  for (int rr = 0; rr < 8; ++rr) {
    float inv = 1.0f / l_i[rr];
    int tok = tb + wave * 16 + rr + 8 * hi;
#pragma unroll
    for (int nt = 0; nt < 8; ++nt)
      out[(size_t)tok * (H_ * DV) + h * DV + nt * 16 + lo] = oacc[nt][rr] * inv;
  }
}

// =====================================================================
// Host-side orchestration
// =====================================================================
extern "C" void kernel_launch(void* const* d_in, const int* in_sizes, int n_in,
                              void* d_out, int out_size, void* d_ws, size_t ws_size,
                              hipStream_t stream) {
  (void)in_sizes; (void)n_in; (void)out_size; (void)ws_size;
  const float* hidden     = (const float*)d_in[0];
  // d_in[1] attention_mask: implicit via causal loop; d_in[2] position_ids: pos = token % S
  const float* wq_a       = (const float*)d_in[3];
  const float* q_a_ln_w   = (const float*)d_in[4];
  const float* wq_b       = (const float*)d_in[5];
  const float* wkv_a      = (const float*)d_in[6];
  const float* kv_a_ln_w  = (const float*)d_in[7];
  const float* wkv_b      = (const float*)d_in[8];
  const float* wo         = (const float*)d_in[9];
  const float* cosb       = (const float*)d_in[10];
  const float* sinb       = (const float*)d_in[11];
  float* out = (float*)d_out;

  // scratch layout (floats)
  float* ws    = (float*)d_ws;
  float* qa    = ws;                                   // [4096, 1536]
  float* qbuf  = qa    + (size_t)TOK * QLORA;          // [4096, 3072]
  float* ckv   = qbuf  + (size_t)TOK * (H_ * DQ);      // [4096, 576]
  float* ckvn  = ckv   + (size_t)TOK * (KVLORA+DROPE); // [4096, 512]
  float* kpe   = ckvn  + (size_t)TOK * KVLORA;         // [4096, 64]
  float* kv    = kpe   + (size_t)TOK * DROPE;          // [4096, 4096]
  float* attnb = kv    + (size_t)TOK * (H_*(DNOPE+DV));// [4096, 2048]

  dim3 blk(128);

  // q_a = hidden @ wq_a.T ; rmsnorm in place
  gemm_wmma_bf16<<<dim3(QLORA/64, TOK/128), blk, 0, stream>>>(hidden, wq_a, qa, TOK, QLORA, HID_);
  rmsnorm_kernel<<<TOK, 256, 0, stream>>>(qa, q_a_ln_w, qa, QLORA, QLORA, QLORA);
  // q = qa_norm @ wq_b.T   -> [tok, h*192]
  gemm_wmma_bf16<<<dim3((H_*DQ)/64, TOK/128), blk, 0, stream>>>(qa, wq_b, qbuf, TOK, H_*DQ, QLORA);
  // ckv = hidden @ wkv_a.T ; rmsnorm first 512 cols -> ckvn
  gemm_wmma_bf16<<<dim3((KVLORA+DROPE)/64, TOK/128), blk, 0, stream>>>(hidden, wkv_a, ckv, TOK, KVLORA+DROPE, HID_);
  rmsnorm_kernel<<<TOK, 256, 0, stream>>>(ckv, kv_a_ln_w, ckvn, KVLORA+DROPE, KVLORA, KVLORA);
  // kv = ckvn @ wkv_b.T    -> [tok, h*256] (k_nope ++ v)
  gemm_wmma_bf16<<<dim3((H_*(DNOPE+DV))/64, TOK/128), blk, 0, stream>>>(ckvn, wkv_b, kv, TOK, H_*(DNOPE+DV), KVLORA);
  // RoPE: q_pe in place, k_pe from ckv tail
  rope_kernel<<<TOK, 256, 0, stream>>>(qbuf, ckv, kpe, cosb, sinb);
  // causal flash attention -> [tok, h*128]
  attn_kernel<<<dim3(S_/QT, B_*H_), blk, 0, stream>>>(qbuf, kv, kpe, attnb);
  // output projection
  gemm_wmma_bf16<<<dim3(HID_/64, TOK/128), blk, 0, stream>>>(attnb, wo, out, TOK, HID_, H_*DV);
}
